// GenesisV2_70875550318622
// MI455X (gfx1250) — matrix operand order repaired and checked by reading, not compiled, not run on hardware
//
#include <hip/hip_runtime.h>
#include <hip/hip_bf16.h>
#include <math.h>

typedef __bf16 bf16;
typedef __attribute__((ext_vector_type(8)))  __bf16 v8bf_t;
typedef __attribute__((ext_vector_type(16))) __bf16 v16bf_t;
typedef __attribute__((ext_vector_type(8)))  float  v8f_t;

union AFrag { v16bf_t v; v8bf_t h[2]; };
union CFrag { v8f_t v; float f[8]; };

__device__ __forceinline__ float gelu_f(float x) {
    return 0.5f * x * (1.0f + erff(x * 0.70710678118654752440f));
}

// ---------------------------------------------------------------------------
// bf16 WMMA GEMM:  C[M,N] = A[M,K] @ B[K,N], B supplied as Bt[N,K].
// Compile-time tiling (no guards in the MAC loop => no exec-mask churn):
//   block = 256 thr = 8 waves; wave grid 4(M) x 2(N);
//   per wave: 16 rows x (NSUB*16) cols; block tile 64 x (NSUB*32).
//   K = KTILES*32, fully unrolled.
// Epilogue: optional bias / exact gelu / residual; fp32 and/or bf16 out.
// ---------------------------------------------------------------------------
template<int NSUB, int KTILES>
__global__ __launch_bounds__(256)
void gemm_bf16_wmma(const bf16* __restrict__ A,  long long sA, int lda,
                    const bf16* __restrict__ Bt, long long sB, int ldb,
                    float* __restrict__ Cf, bf16* __restrict__ Cb,
                    long long sC, int ldc,
                    const float* __restrict__ bias,
                    const float* __restrict__ Res, long long sR,
                    int ops)
{
    const int bz = blockIdx.z;
    A  += (long long)bz * sA;
    Bt += (long long)bz * sB;
    const long long cbase = (long long)bz * sC;

    const int w    = threadIdx.x >> 5;
    const int lane = threadIdx.x & 31;
    const int wm   = w & 3;                       // 4 row-tiles of 16
    const int wn   = w >> 2;                      // 2 col-slabs
    const int row0 = blockIdx.x * 64 + wm * 16;
    const int col0 = blockIdx.y * (NSUB * 32) + wn * (NSUB * 16);

    CFrag acc[NSUB];
#pragma unroll
    for (int t = 0; t < NSUB; ++t)
#pragma unroll
        for (int r = 0; r < 8; ++r) acc[t].f[r] = 0.0f;

    // A fragment (16-bit 16x32 layout): lane<16 -> K {0..7,16..23},
    // lane>=16 -> K {8..15,24..31}; (lane&15) = M row.
    const bf16* aptr = A + (long long)(row0 + (lane & 15)) * lda + ((lane >> 4) * 8);
    // B fragment from Bt[N,K]: (lane&15) = column; same K layout as A.
    const int boff = (lane >> 4) * 8;
    const bf16* bptr[NSUB];
#pragma unroll
    for (int t = 0; t < NSUB; ++t)
        bptr[t] = Bt + (long long)(col0 + t * 16 + (lane & 15)) * ldb + boff;

#pragma unroll
    for (int kt = 0; kt < KTILES; ++kt) {
        const int kk = kt * 32;
        AFrag a;
        a.h[0] = *(const v8bf_t*)(aptr + kk);
        a.h[1] = *(const v8bf_t*)(aptr + kk + 16);
        AFrag bfr[NSUB];
#pragma unroll
        for (int t = 0; t < NSUB; ++t) {
            bfr[t].h[0] = *(const v8bf_t*)(bptr[t] + kk);
            bfr[t].h[1] = *(const v8bf_t*)(bptr[t] + kk + 16);
        }
#pragma unroll
        for (int t = 0; t < NSUB; ++t)
            acc[t].v = __builtin_amdgcn_wmma_f32_16x16x32_bf16(
                false, a.v, false, bfr[t].v, (short)0, acc[t].v, false, false);
    }

    // Epilogue.  C/D layout: VGPR r -> (M = r + 8*(lane>>4), N = lane&15).
    const int rlo  = (lane >> 4) * 8;
    const int ccol = lane & 15;
#pragma unroll
    for (int t = 0; t < NSUB; ++t) {
        const int col = col0 + t * 16 + ccol;
        const float bv = (ops & 1) ? bias[col] : 0.0f;
#pragma unroll
        for (int r = 0; r < 8; ++r) {
            const int row = row0 + rlo + r;
            float v = acc[t].f[r] + bv;
            if (ops & 2) v = gelu_f(v);
            if (ops & 4) v += Res[(long long)bz * sR + (long long)row * ldc + col];
            const long long ci = cbase + (long long)row * ldc + col;
            if (Cf) Cf[ci] = v;
            if (Cb) Cb[ci] = (bf16)v;
        }
    }
}

// ---------------------------------------------------------------------------
// Small helper kernels
// ---------------------------------------------------------------------------
__global__ void copy_cvt(const float* __restrict__ src, float* dstf,
                         bf16* __restrict__ dstb, int n)
{
    int i = blockIdx.x * 256 + threadIdx.x;
    if (i >= n) return;
    float v = src[i];
    if (dstf) dstf[i] = v;
    dstb[i] = (bf16)v;
}

// Wt[n*K + k] = W[k*N + n] * scale   (W is K x N row-major)
__global__ void prep_wT(const float* __restrict__ W, bf16* __restrict__ Wt,
                        int K, int N, float scale)
{
    int i = blockIdx.x * 256 + threadIdx.x;
    if (i >= K * N) return;
    int n = i / K, k = i - n * K;
    Wt[i] = (bf16)(W[k * N + n] * scale);
}

// Split Wv1 (2D x D) into (top+bot)^T and bot^T, bf16.
__global__ void prep_wv1(const float* __restrict__ Wv1, bf16* __restrict__ WsT,
                         bf16* __restrict__ WbT)
{
    int i = blockIdx.x * 256 + threadIdx.x;
    if (i >= 128 * 128) return;
    int n = i / 128, k = i - n * 128;
    float top = Wv1[k * 128 + n];
    float bot = Wv1[(128 + k) * 128 + n];
    WsT[i] = (bf16)(top + bot);
    WbT[i] = (bf16)bot;
}

// Row softmax over 64 elements, in place.
__global__ void softmax64(float* __restrict__ s)
{
    int row = blockIdx.x, t = threadIdx.x;
    __shared__ float buf[64];
    float v = s[row * 64 + t];
    buf[t] = v; __syncthreads();
    for (int o = 32; o > 0; o >>= 1) { if (t < o) buf[t] = fmaxf(buf[t], buf[t + o]); __syncthreads(); }
    float m = buf[0]; __syncthreads();
    float e = expf(v - m);
    buf[t] = e; __syncthreads();
    for (int o = 32; o > 0; o >>= 1) { if (t < o) buf[t] += buf[t + o]; __syncthreads(); }
    s[row * 64 + t] = e / buf[0];
}

// g[b,i,ch] = sum_j attn[b,i,j] * gelu(c[b,i,ch] - d[b,j,ch])   -> bf16
__global__ void pairwise_reduce(const float* __restrict__ c,
                                const float* __restrict__ d,
                                const float* __restrict__ attn,
                                bf16* __restrict__ g)
{
    int i = blockIdx.x, b = blockIdx.y, ch = threadIdx.x;
    __shared__ float w[64];
    if (ch < 64) w[ch] = attn[((b * 64 + i) * 64) + ch];
    __syncthreads();
    float ci  = c[((long long)(b * 64 + i)) * 128 + ch];
    const float* dp = d + (long long)b * 64 * 128 + ch;
    float acc = 0.0f;
#pragma unroll 4
    for (int j = 0; j < 64; ++j)
        acc += w[j] * gelu_f(ci - dp[j * 128]);
    g[((long long)(b * 64 + i)) * 128 + ch] = (bf16)acc;
}

__global__ void layernorm_rows(const float* __restrict__ h,
                               const float* __restrict__ g,
                               const float* __restrict__ b,
                               bf16* __restrict__ out)
{
    int row = blockIdx.x, t = threadIdx.x;   // 128 threads
    __shared__ float buf[128];
    float v = h[(long long)row * 128 + t];
    buf[t] = v; __syncthreads();
    for (int o = 64; o > 0; o >>= 1) { if (t < o) buf[t] += buf[t + o]; __syncthreads(); }
    float mu = buf[0] * (1.0f / 128.0f); __syncthreads();
    float dv = v - mu;
    buf[t] = dv * dv; __syncthreads();
    for (int o = 64; o > 0; o >>= 1) { if (t < o) buf[t] += buf[t + o]; __syncthreads(); }
    float var = buf[0] * (1.0f / 128.0f);
    float r = rsqrtf(var + 1e-5f);
    out[(long long)row * 128 + t] = (bf16)(dv * r * g[t] + b[t]);
}

__global__ void pool_sum(const float* __restrict__ x, bf16* __restrict__ out)
{
    int b = blockIdx.x, t = threadIdx.x;     // 128 threads
    float s = 0.0f;
    for (int i = 0; i < 64; ++i) s += x[((long long)(b * 64 + i)) * 128 + t];
    out[b * 128 + t] = (bf16)s;
}

// ---------------------------------------------------------------------------
// Host orchestration
// ---------------------------------------------------------------------------
extern "C" void kernel_launch(void* const* d_in, const int* in_sizes, int n_in,
                              void* d_out, int out_size, void* d_ws, size_t ws_size,
                              hipStream_t stream)
{
    constexpr int Bn = 128, Kt = 64, D = 128, S = 128, Ll = 2;
    constexpr int NT = Bn * Kt;          // 8192 tokens
    constexpr int DD = D * D;            // 16384

    const float* x_in = (const float*)d_in[0];
    const float* Wk   = (const float*)d_in[1];
    const float* Wq   = (const float*)d_in[2];
    const float* Wv1  = (const float*)d_in[3];
    const float* bv1  = (const float*)d_in[4];
    const float* Wv2  = (const float*)d_in[5];
    const float* bv2  = (const float*)d_in[6];
    const float* lng  = (const float*)d_in[7];
    const float* lnb_ = (const float*)d_in[8];
    const float* Wm1  = (const float*)d_in[9];
    const float* bm1  = (const float*)d_in[10];
    const float* Wm2  = (const float*)d_in[11];
    const float* bm2  = (const float*)d_in[12];
    const float* Wp   = (const float*)d_in[13];
    const float* bp   = (const float*)d_in[14];
    const float* Wmu  = (const float*)d_in[15];
    const float* bmu  = (const float*)d_in[16];
    const float* Wsp  = (const float*)d_in[17];
    const float* bsp  = (const float*)d_in[18];
    float* out = (float*)d_out;

    // ---- workspace carve-out ----
    char* base = (char*)d_ws;
    size_t off = 0;
    auto al = [&](size_t bytes) -> char* {
        char* p = base + off;
        off = (off + bytes + 255) & ~(size_t)255;
        return p;
    };
    float* xcur   = (float*)al((size_t)NT * D * 4);
    float* h2     = (float*)al((size_t)NT * D * 4);
    float* cbuf   = (float*)al((size_t)NT * D * 4);
    float* dbuf   = (float*)al((size_t)NT * D * 4);
    float* scores = (float*)al((size_t)Bn * Kt * Kt * 4);
    bf16*  xb     = (bf16*)al((size_t)NT * D * 2);
    bf16*  kb     = (bf16*)al((size_t)NT * D * 2);
    bf16*  qb     = (bf16*)al((size_t)NT * D * 2);
    bf16*  gb     = (bf16*)al((size_t)NT * D * 2);
    bf16*  lnbuf  = (bf16*)al((size_t)NT * D * 2);
    bf16*  m1b    = (bf16*)al((size_t)NT * D * 2);
    bf16*  pooledb= (bf16*)al((size_t)Bn * D * 2);
    bf16*  projb  = (bf16*)al((size_t)Bn * S * 2);
    bf16*  wkT    = (bf16*)al((size_t)Ll * DD * 2);
    bf16*  wqT    = (bf16*)al((size_t)Ll * DD * 2);
    bf16*  w1sT   = (bf16*)al((size_t)Ll * DD * 2);
    bf16*  w1bT   = (bf16*)al((size_t)Ll * DD * 2);
    bf16*  w2T    = (bf16*)al((size_t)Ll * DD * 2);
    bf16*  wm1T   = (bf16*)al((size_t)Ll * DD * 2);
    bf16*  wm2T   = (bf16*)al((size_t)Ll * DD * 2);
    bf16*  wpT    = (bf16*)al((size_t)DD * 2);
    bf16*  wmuT   = (bf16*)al((size_t)S * S * 2);
    bf16*  wspT   = (bf16*)al((size_t)S * S * 2);
    (void)ws_size; (void)in_sizes; (void)n_in; (void)out_size;

    const float scale = 0.08838834764831845f;   // D^-0.5
    const dim3 blk256(256);
    const dim3 gW((DD + 255) / 256);

    // ---- weight prep (bf16, transposed; fold q-scale; split Wv1) ----
    for (int l = 0; l < Ll; ++l) {
        prep_wT <<<gW, blk256, 0, stream>>>(Wk  + l * DD, wkT  + l * DD, D, D, 1.0f);
        prep_wT <<<gW, blk256, 0, stream>>>(Wq  + l * DD, wqT  + l * DD, D, D, scale);
        prep_wv1<<<gW, blk256, 0, stream>>>(Wv1 + l * 2 * DD, w1sT + l * DD, w1bT + l * DD);
        prep_wT <<<gW, blk256, 0, stream>>>(Wv2 + l * DD, w2T  + l * DD, D, D, 1.0f);
        prep_wT <<<gW, blk256, 0, stream>>>(Wm1 + l * DD, wm1T + l * DD, D, D, 1.0f);
        prep_wT <<<gW, blk256, 0, stream>>>(Wm2 + l * DD, wm2T + l * DD, D, D, 1.0f);
    }
    prep_wT<<<gW, blk256, 0, stream>>>(Wp,  wpT,  D, S, 1.0f);
    prep_wT<<<gW, blk256, 0, stream>>>(Wmu, wmuT, S, S, 1.0f);
    prep_wT<<<gW, blk256, 0, stream>>>(Wsp, wspT, S, S, 1.0f);

    // ---- x -> xcur (fp32) + xb (bf16) ----
    copy_cvt<<<dim3((NT * D + 255) / 256), blk256, 0, stream>>>(x_in, xcur, xb, NT * D);

    const dim3 gMain(NT / 64, 1, 1);            // 8192 x 128 x 128 GEMMs
    const dim3 gAttn(1, 1, Bn);                 // batched 64 x 64 x 128
    const dim3 gSmall(2, 1, 1);                 // 128 x 128 x 128 GEMMs

    for (int l = 0; l < Ll; ++l) {
        if (l) copy_cvt<<<dim3((NT * D + 255) / 256), blk256, 0, stream>>>(xcur, nullptr, xb, NT * D);

        // k, q (bf16 only; q-scale folded into weights)
        gemm_bf16_wmma<4,4><<<gMain, blk256, 0, stream>>>(xb, 0, D, wkT + l * DD, 0, D,
            nullptr, kb, 0, D, nullptr, nullptr, 0, 0);
        gemm_bf16_wmma<4,4><<<gMain, blk256, 0, stream>>>(xb, 0, D, wqT + l * DD, 0, D,
            nullptr, qb, 0, D, nullptr, nullptr, 0, 0);
        // c = x@(W1top+W1bot) + bv1 ; d = x@W1bot     (pairwise MLP linearized)
        gemm_bf16_wmma<4,4><<<gMain, blk256, 0, stream>>>(xb, 0, D, w1sT + l * DD, 0, D,
            cbuf, nullptr, 0, D, bv1 + l * D, nullptr, 0, 1);
        gemm_bf16_wmma<4,4><<<gMain, blk256, 0, stream>>>(xb, 0, D, w1bT + l * DD, 0, D,
            dbuf, nullptr, 0, D, nullptr, nullptr, 0, 0);
        // attention scores: per batch q[64,128] @ k[64,128]^T (k rows already = Bt rows)
        gemm_bf16_wmma<2,4><<<gAttn, blk256, 0, stream>>>(qb, (long long)Kt * D, D,
            kb, (long long)Kt * D, D, scores, nullptr, (long long)Kt * Kt, Kt,
            nullptr, nullptr, 0, 0);
        softmax64<<<dim3(NT), dim3(64), 0, stream>>>(scores);
        // g_i = sum_j attn_ij * gelu(c_i - d_j)
        pairwise_reduce<<<dim3(Kt, Bn), dim3(D), 0, stream>>>(cbuf, dbuf, scores, gb);
        // h2 = x + g @ Wv2 + bv2
        gemm_bf16_wmma<4,4><<<gMain, blk256, 0, stream>>>(gb, 0, D, w2T + l * DD, 0, D,
            h2, nullptr, 0, D, bv2 + l * D, xcur, 0, 1 | 4);
        // layernorm -> bf16
        layernorm_rows<<<dim3(NT), dim3(D), 0, stream>>>(h2, lng + l * D, lnb_ + l * D, lnbuf);
        // mlp: gelu(ln @ Wm1 + bm1) -> bf16
        gemm_bf16_wmma<4,4><<<gMain, blk256, 0, stream>>>(lnbuf, 0, D, wm1T + l * DD, 0, D,
            nullptr, m1b, 0, D, bm1 + l * D, nullptr, 0, 1 | 2);
        // x_new = h2 + m1 @ Wm2 + bm2
        gemm_bf16_wmma<4,4><<<gMain, blk256, 0, stream>>>(m1b, 0, D, wm2T + l * DD, 0, D,
            xcur, nullptr, 0, D, bm2 + l * D, h2, 0, 1 | 4);
    }

    // ---- head ----
    pool_sum<<<dim3(Bn), dim3(D), 0, stream>>>(xcur, pooledb);
    gemm_bf16_wmma<4,4><<<gSmall, blk256, 0, stream>>>(pooledb, 0, D, wpT, 0, D,
        nullptr, projb, 0, S, bp, nullptr, 0, 1 | 2);
    gemm_bf16_wmma<4,4><<<gSmall, blk256, 0, stream>>>(projb, 0, S, wmuT, 0, S,
        out, nullptr, 0, S, bmu, nullptr, 0, 1);
    gemm_bf16_wmma<4,4><<<gSmall, blk256, 0, stream>>>(projb, 0, S, wspT, 0, S,
        out + Bn * S, nullptr, 0, S, bsp, nullptr, 0, 1);
}
